// LSTM_noforget_64501818851785
// MI455X (gfx1250) — matrix operand order repaired and checked by reading, not compiled, not run on hardware
//
#include <hip/hip_runtime.h>
#include <stdint.h>

// ---------------------------------------------------------------------------
// LSTM (no forget gate), B=64 S=2048 I=128 H=512 O=128, MI455X / gfx1250.
// Persistent fused recurrence with bf16 WMMA (v_wmma_f32_16x16x32_bf16).
//  - x pre-swizzled to WMMA A-tile bf16 layout (once) -> hot loop has zero cvt
//  - h ping-pong buffers kept directly in A-tile bf16 layout
//  - U B-tiles resident in LDS (48KB), W B-tiles in L2-resident staged buffer
//  - per-step grid barrier (atomic counter + s_sleep backoff)
// ---------------------------------------------------------------------------

typedef __attribute__((ext_vector_type(16))) __bf16 v16bf;
typedef __attribute__((ext_vector_type(8)))  float  v8f;

#define NWG 32                 // persistent workgroups (one 16-col slice of H each)
#define WAVES_PER_WG 12        // 3 gates x 4 row-tiles
#define TPB (WAVES_PER_WG * 32)

constexpr int B_ = 64, S_ = 2048, I_ = 128, H_ = 512, O_ = 128;

// d_out layout (float elements): out_seq | hidden_seq | h_T | c_T | out_T
constexpr size_t OFF_HID  = (size_t)B_ * S_ * O_;            // 16777216
constexpr size_t OFF_HT   = OFF_HID + (size_t)B_ * S_ * H_;  // 83886080
constexpr size_t OFF_CT   = OFF_HT  + (size_t)B_ * H_;
constexpr size_t OFF_OUTT = OFF_CT  + (size_t)B_ * H_;

// ws layout (bytes)
constexpr size_t WS_CNT  = 0;                                  // grid-barrier counter
constexpr size_t WS_HBUF = 256;                                // 2 x 64 A-tiles bf16 = 131072 B
constexpr size_t WS_W    = WS_HBUF + 131072;                   // staged W  : 4x96 tiles
constexpr size_t WS_U    = WS_W + (size_t)4  * 96 * 256 * 4;   // staged U  : 16x96 tiles
constexpr size_t WS_WO   = WS_U + (size_t)16 * 96 * 256 * 4;   // staged wo : 16x8 tiles
constexpr size_t WS_X    = WS_WO + (size_t)16 * 8 * 256 * 4;   // staged x  : S*4*4 A-tiles = 32MB

union BF16x16 { v16bf v; uint32_t u[8]; uint4 q[2]; };

// Round-to-nearest fp32->bf16 pair pack: 2 adds + 1 v_perm_b32.
__device__ inline uint32_t pack2bf(float a, float b) {
  uint32_t ua = __float_as_uint(a) + 0x8000u;
  uint32_t ub = __float_as_uint(b) + 0x8000u;
  return __builtin_amdgcn_perm(ub, ua, 0x07060302u);
}
__device__ inline uint16_t bf16r(float a) {
  return (uint16_t)((__float_as_uint(a) + 0x8000u) >> 16);
}
__device__ inline float sigmoidf_(float v) { return 1.0f / (1.0f + __expf(-v)); }
__device__ inline float tanhf_(float v) {
  float a = fabsf(v);
  float e = __expf(-2.0f * a);
  float r = (1.0f - e) / (1.0f + e);
  return copysignf(r, v);
}

// Load one pre-swizzled 256-dword tile operand: 2 x b128 per lane.
__device__ inline v16bf load_tile_g(const uint32_t* tile_base, int lane) {
  const uint4* p = (const uint4*)(tile_base + lane * 8);
  BF16x16 t; t.q[0] = p[0]; t.q[1] = p[1];
  return t.v;
}
__device__ inline v16bf load_tile_lds(const uint32_t* base, int slot, int lane) {
  const uint4* p = (const uint4*)(base + slot * 256 + lane * 8);
  BF16x16 t; t.q[0] = p[0]; t.q[1] = p[1];
  return t.v;
}

// Build A operand (16x32 bf16) from an fp32 row pointer (row = this lane's M),
// using float4 loads (k runs are contiguous within each 4-dword group).
__device__ inline v16bf load_a_f32(const float* rowptr, int ktBase, int lane) {
  BF16x16 t;
  const int hi = (lane >> 4) << 3;
#pragma unroll
  for (int g = 0; g < 2; g++) {
    int kb = ktBase + g * 16 + hi;
    float4 f0 = *(const float4*)(rowptr + kb);
    float4 f1 = *(const float4*)(rowptr + kb + 4);
    t.u[g * 4 + 0] = pack2bf(f0.x, f0.y);
    t.u[g * 4 + 1] = pack2bf(f0.z, f0.w);
    t.u[g * 4 + 2] = pack2bf(f1.x, f1.y);
    t.u[g * 4 + 3] = pack2bf(f1.z, f1.w);
  }
  return t.v;
}

// ---------------------------------------------------------------------------
__global__ void init_kernel(uint32_t* cnt, uint32_t* hbuf) {
  int gid = blockIdx.x * blockDim.x + threadIdx.x;
  if (gid < 32768) hbuf[gid] = 0u;       // both ping-pong banks -> h0 = 0
  if (gid < 64)    cnt[gid]  = 0u;       // barrier counter
}

// Swizzle an fp32 KxN matrix into bf16 WMMA B-operand tiles.
__device__ inline void stage_b(const float* src, int N, int numNt, uint32_t* dst, int idx) {
  int j    = idx & 7;
  int lane = (idx >> 3) & 31;
  int tile = idx >> 8;
  int nt   = tile % numNt;
  int kt   = tile / numNt;
  int n    = nt * 16 + (lane & 15);
  int k0   = kt * 32 + ((j < 4) ? 0 : 16) + ((lane >> 4) << 3) + 2 * (j & 3);
  dst[idx] = pack2bf(src[(size_t)k0 * N + n], src[(size_t)(k0 + 1) * N + n]);
}

// Swizzle x (B x S x I fp32) into bf16 WMMA A-tiles: tile = t*16 + mtile*4 + kt.
__device__ inline void stage_x(const float* x, uint32_t* dst, int idx) {
  int j    = idx & 7;
  int lane = (idx >> 3) & 31;
  int tile = idx >> 8;
  int kt   = tile & 3;
  int mt   = (tile >> 2) & 3;
  int t    = tile >> 4;
  int m    = mt * 16 + (lane & 15);
  int k0   = kt * 32 + ((j < 4) ? 0 : 16) + ((lane >> 4) << 3) + 2 * (j & 3);
  const float* p = x + ((size_t)m * S_ + t) * I_ + k0;
  float2 f = *(const float2*)p;
  dst[idx] = pack2bf(f.x, f.y);
}

__global__ void stage_kernel(const float* x, const float* W, const float* U, const float* wo,
                             uint32_t* dX, uint32_t* dW, uint32_t* dU, uint32_t* dWO) {
  int gid = blockIdx.x * blockDim.x + threadIdx.x;
  if (gid < 98304)               stage_b(W,  1536, 96, dW,  gid);
  else if (gid < 491520)         stage_b(U,  1536, 96, dU,  gid - 98304);
  else if (gid < 524288)         stage_b(wo, 128,  8,  dWO, gid - 491520);
  else if (gid < 524288 + 8388608) stage_x(x, dX, gid - 524288);
}

// ---------------------------------------------------------------------------
// Persistent recurrent kernel. 32 workgroups; wg owns h-columns [wg*16, wg*16+16).
__global__ void __launch_bounds__(TPB, 1)
lstm_persistent(const uint32_t* __restrict__ sX, const float* __restrict__ bias,
                const uint32_t* __restrict__ sW, const uint32_t* __restrict__ sU,
                uint32_t* hbuf, uint32_t* cnt, float* out) {
  __shared__ uint32_t uTiles[48 * 256];  // this wg's U B-tiles: slot = gate*16 + kt  (48KB)
  __shared__ float    gates[3 * 64 * 16];

  const int tid   = threadIdx.x;
  const int lane  = tid & 31;
  const int wid   = tid >> 5;        // 0..11
  const int gate  = wid >> 2;        // 0..2  (i, g, o)
  const int mtile = wid & 3;         // 0..3
  const int wg    = blockIdx.x;      // 0..31
  const int j0    = wg * 16;

  // Stage this wg's U B-tiles into LDS (48 tiles, 4 per wave).
#pragma unroll
  for (int s = 0; s < 4; s++) {
    int slot = wid * 4 + s;                    // 0..47
    int g = slot >> 4, kt = slot & 15;
    int ntg = g * 32 + wg;
    const uint4* src = (const uint4*)(sU + (((size_t)kt * 96 + ntg) * 32 + lane) * 8);
    uint4* dst = (uint4*)(uTiles + (size_t)slot * 256 + lane * 8);
    dst[0] = src[0]; dst[1] = src[1];
  }
  __syncthreads();

  const int   ntg   = gate * 32 + wg;
  const float bval  = bias[gate * H_ + j0 + (lane & 15)];

  // Loop-invariant pieces of the swizzled h write (col fixed per lane).
  const int nloc  = lane & 15;
  const int col   = j0 + nloc;
  const int hkt   = col >> 5;
  const int kk    = col & 31;
  const int hhalf = ((kk >> 3) & 1) << 4;
  const int hj    = ((kk & 7) >> 1) + ((kk >= 16) ? 4 : 0);
  const int hpos  = kk & 1;

  float* outHid = out + OFF_HID;

  for (int t = 0; t < S_; t++) {
    const uint32_t* hbank  = hbuf + ((t & 1) * 16384);
    uint16_t*       hw16   = (uint16_t*)(hbuf + (((t + 1) & 1) * 16384));
    const uint32_t* xtiles = sX + (((size_t)t * 4 + mtile) * 4) * 256;
    const uint32_t* htiles = hbank + (size_t)mtile * 16 * 256;

    v8f acc = {};
    // x_t @ W   (K = 128 -> 4 WMMAs); B from staged global (L2-resident)
#pragma unroll
    for (int kt = 0; kt < 4; kt++) {
      v16bf a = load_tile_g(xtiles + kt * 256, lane);
      v16bf b = load_tile_g(sW + (((size_t)kt * 96 + ntg) * 32) * 8, lane);
      acc = __builtin_amdgcn_wmma_f32_16x16x32_bf16(false, a, false, b, (short)0, acc, false, false);
    }
    // h_{t-1} @ U  (K = 512 -> 16 WMMAs); B from LDS
#pragma unroll
    for (int kt = 0; kt < 16; kt++) {
      v16bf a = load_tile_g(htiles + kt * 256, lane);
      v16bf b = load_tile_lds(uTiles, gate * 16 + kt, lane);
      acc = __builtin_amdgcn_wmma_f32_16x16x32_bf16(false, a, false, b, (short)0, acc, false, false);
    }
    if (t + 1 < S_) __builtin_prefetch(xtiles + 16 * 256, 0, 1);  // next step's x tiles

    // Publish gate tile to LDS (C layout: VGPR r -> M = r + 8*(lane>=16), N = lane&15)
#pragma unroll
    for (int r = 0; r < 8; r++) {
      int m = mtile * 16 + r + ((lane >> 4) << 3);
      gates[(gate * 64 + m) * 16 + nloc] = acc[r] + bval;
    }
    __syncthreads();

    // Gate-0 waves combine i/g/o -> c, h ; write hidden_seq + swizzled h.
    if (gate == 0) {
#pragma unroll
      for (int r = 0; r < 8; r++) {
        int m = mtile * 16 + r + ((lane >> 4) << 3);   // batch row
        float iv = sigmoidf_(gates[(0 * 64 + m) * 16 + nloc]);
        float gv = tanhf_  (gates[(1 * 64 + m) * 16 + nloc]);
        float ov = sigmoidf_(gates[(2 * 64 + m) * 16 + nloc]);
        float c = iv * gv;
        float h = ov * tanhf_(c);
        outHid[((size_t)m * S_ + t) * H_ + col] = h;              // hidden_seq fp32
        // h in A-tile swizzled bf16: tile = (m>>4)*16 + hkt
        int lane_t = (m & 15) + hhalf;
        size_t u16idx = ((((size_t)(m >> 4) * 16 + hkt) * 32 + lane_t) * 8 + hj) * 2 + hpos;
        hw16[u16idx] = bf16r(h);
        if (t == S_ - 1) {
          out[OFF_HT + (size_t)m * H_ + col] = h;
          out[OFF_CT + (size_t)m * H_ + col] = c;
        }
      }
    }

    // Grid-wide barrier before next step's h reads.
    __syncthreads();
    if (tid == 0) {
      __threadfence();
      atomicAdd(cnt, 1u);
      unsigned target = (unsigned)(t + 1) * NWG;
      while (atomicAdd(cnt, 0u) < target) __builtin_amdgcn_s_sleep(1);
      __threadfence();
    }
    __syncthreads();
  }
}

// ---------------------------------------------------------------------------
// out_seq = hidden_seq @ wo ; also peels out_T at s == S-1.
__global__ void __launch_bounds__(256)
proj_kernel(const uint32_t* __restrict__ sWO, float* out) {
  const int lane = threadIdx.x & 31;
  const int id   = blockIdx.x * 8 + (threadIdx.x >> 5);   // 65536 tiles total
  const int mt   = id >> 3;                               // B*S/16 row tiles
  const int ntl  = id & 7;                                // O/16 col tiles

  const float* hid  = out + OFF_HID;
  const int    mrow = mt * 16 + (lane & 15);
  const float* arow = hid + (size_t)mrow * H_;

  v8f acc = {};
#pragma unroll
  for (int kt = 0; kt < 16; kt++) {
    v16bf a = load_a_f32(arow, kt * 32, lane);
    v16bf b = load_tile_g(sWO + (((size_t)kt * 8 + ntl) * 32) * 8, lane);
    acc = __builtin_amdgcn_wmma_f32_16x16x32_bf16(false, a, false, b, (short)0, acc, false, false);
  }
#pragma unroll
  for (int r = 0; r < 8; r++) {
    int m = mt * 16 + r + ((lane >> 4) << 3);             // global row = b*S + s
    int n = ntl * 16 + (lane & 15);
    float v = acc[r];
    out[(size_t)m * O_ + n] = v;                          // out_seq
    if ((m & (S_ - 1)) == (S_ - 1))                       // s == S-1
      out[OFF_OUTT + (size_t)(m >> 11) * O_ + n] = v;     // out_T
  }
}

// ---------------------------------------------------------------------------
extern "C" void kernel_launch(void* const* d_in, const int* in_sizes, int n_in,
                              void* d_out, int out_size, void* d_ws, size_t ws_size,
                              hipStream_t stream) {
  const float* x    = (const float*)d_in[0];
  const float* W    = (const float*)d_in[1];
  const float* U    = (const float*)d_in[2];
  const float* bias = (const float*)d_in[3];
  const float* wo   = (const float*)d_in[4];
  float* out = (float*)d_out;

  char* ws = (char*)d_ws;
  uint32_t* cnt  = (uint32_t*)(ws + WS_CNT);
  uint32_t* hbuf = (uint32_t*)(ws + WS_HBUF);
  uint32_t* sW   = (uint32_t*)(ws + WS_W);
  uint32_t* sU   = (uint32_t*)(ws + WS_U);
  uint32_t* sWO  = (uint32_t*)(ws + WS_WO);
  uint32_t* sX   = (uint32_t*)(ws + WS_X);

  hipLaunchKernelGGL(init_kernel,  dim3(128),   dim3(256), 0, stream, cnt, hbuf);
  hipLaunchKernelGGL(stage_kernel, dim3(34816), dim3(256), 0, stream,
                     x, W, U, wo, sX, sW, sU, sWO);
  hipLaunchKernelGGL(lstm_persistent, dim3(NWG), dim3(TPB), 0, stream,
                     sX, bias, sW, sU, hbuf, cnt, out);
  hipLaunchKernelGGL(proj_kernel,  dim3(8192),  dim3(256), 0, stream, sWO, out);
}
